// SymptomStream_64338610095075
// MI455X (gfx1250) — compile-verified, matrix operand order checked by reading, not run on hardware
//
#include <hip/hip_runtime.h>
#include <hip/hip_bf16.h>

typedef __attribute__((ext_vector_type(16))) _Float16 v16h;
typedef __attribute__((ext_vector_type(8)))  _Float16 v8h;
typedef __attribute__((ext_vector_type(8)))  float    v8f;

#define NLAY 6
#define DIM  38
#define SEQ  64
#define HDIM 19
#define N3   114   // 3*DIM
#define NK   8

// ---- workspace layout (in _Float16 elements) ----
// WQKV : NLAY x 128 x 64   (N=114 padded->128 rows, K=38 padded->64, zero pad)
// WOUT : NLAY x  64 x 64
// WL1  : NLAY x  64 x 64
// WL2  : NLAY x  64 x 64
#define WS_QKV   0
#define WS_OUT   49152
#define WS_L1    73728
#define WS_L2    98304
#define WS_TOTAL 122880

// ---- LDS layout (byte offsets) ----
#define SM_XF     0        // f32 [64][40]                     10240
#define SM_XH     10240    // f16 [64][72]                      9216  (zero init)
#define SM_QBUF   19456    // f16 [2][64][40]                  10240  (zero init)
#define SM_KBUF   29696    // f16 [2][64][40]                  10240  (zero init)
#define SM_VT     39936    // f16 [2][32][72]  (transposed V)   9216  (zero init)
#define SM_AOB    49152    // f16 [64][72]                      9216  (zero init)
#define SM_WQKV   58368    // f16 [128][72]                    18432
#define SM_WSM    76800    // f16 3 x [64][72]                 27648
#define SM_PBUF   104448   // f16 [2][64][72]  (probs / ffn h) 18432
#define SM_SBUF   122880   // f32 [2][64][66]  (scores/out/pool)33792
#define SM_TOTAL  156672

__device__ __forceinline__ v16h load_frag(const _Float16* p) {
    union { v16h v; v8h h[2]; } u;
    u.h[0] = *(const v8h*)(p);
    u.h[1] = *(const v8h*)(p + 8);
    return u.v;
}

__device__ __forceinline__ v8f wmma16(v16h a, v16h b, v8f c) {
    return __builtin_amdgcn_wmma_f32_16x16x32_f16(
        false, a, false, b, (short)0, c, false, false);
}

// ------------------------------------------------------------------
// Weight conversion: fp32 -> zero-padded f16 B-operand layout in d_ws
// ------------------------------------------------------------------
__global__ void cvt_weights_kernel(const float* __restrict__ qkvw,
                                   const float* __restrict__ outw,
                                   const float* __restrict__ l1w,
                                   const float* __restrict__ l2w,
                                   _Float16* __restrict__ ws) {
    int idx = blockIdx.x * blockDim.x + threadIdx.x;
    if (idx >= WS_TOTAL) return;
    if (idx < WS_OUT) {
        int l = idx / 8192, r = idx % 8192, n = r >> 6, k = r & 63;
        float v = (n < N3 && k < DIM) ? qkvw[(l * N3 + n) * DIM + k] : 0.f;
        ws[idx] = (_Float16)v;
    } else {
        int j = idx - WS_OUT;
        int which = j / 24576; j -= which * 24576;
        int l = j / 4096, r = j % 4096, n = r >> 6, k = r & 63;
        const float* W = (which == 0) ? outw : (which == 1 ? l1w : l2w);
        float v = (n < DIM && k < DIM) ? W[(l * DIM + n) * DIM + k] : 0.f;
        ws[idx] = (_Float16)v;
    }
}

// ------------------------------------------------------------------
// Fused per-sequence transformer: one block per batch element
// ------------------------------------------------------------------
__global__ __launch_bounds__(256)
void symptom_fused_kernel(const float* __restrict__ x,
                          const float* __restrict__ ipb,
                          const float* __restrict__ outb,
                          const float* __restrict__ l1b,
                          const float* __restrict__ l2b,
                          const float* __restrict__ ln1w,
                          const float* __restrict__ ln1b,
                          const float* __restrict__ ln2w,
                          const float* __restrict__ ln2b,
                          const float* __restrict__ attnw,
                          const float* __restrict__ attnb,
                          const float* __restrict__ clfw,
                          const float* __restrict__ clfb,
                          const _Float16* __restrict__ ws,
                          float* __restrict__ out) {
    __shared__ __attribute__((aligned(16))) char smem[SM_TOTAL];

    float*    xf   = (float*)   (smem + SM_XF);    // [64][40]
    _Float16* xh   = (_Float16*)(smem + SM_XH);    // [64][72]
    _Float16* qbuf = (_Float16*)(smem + SM_QBUF);  // [2][64][40]
    _Float16* kbuf = (_Float16*)(smem + SM_KBUF);  // [2][64][40]
    _Float16* vT   = (_Float16*)(smem + SM_VT);    // [2][32][72]
    _Float16* aob  = (_Float16*)(smem + SM_AOB);   // [64][72]
    _Float16* wq   = (_Float16*)(smem + SM_WQKV);  // [128][72]
    _Float16* wsm  = (_Float16*)(smem + SM_WSM);   // 3 x [64][72]
    _Float16* pbuf = (_Float16*)(smem + SM_PBUF);  // [2][64][72]
    float*    sbuf = (float*)   (smem + SM_SBUF);  // [2][64][66]

    const int tid  = threadIdx.x;
    const int lane = tid & 31;
    const int wave = tid >> 5;
    const int b    = blockIdx.x;
    const int lhi  = (lane >> 4) << 3;   // +8 rows for upper half-wave
    const int lcol = lane & 15;
    const int lkof = (lane >> 4) * 16;   // K offset for fragment loads

    // ---- zero the f16 pad-sensitive region (XH..AOB is contiguous) ----
    {
        int4* z = (int4*)(smem + SM_XH);
        const int nchunk = (SM_WQKV - SM_XH) / 16;     // 3008
        for (int c = tid; c < nchunk; c += 256) z[c] = int4{0, 0, 0, 0};
    }
    __syncthreads();

    // ---- load x[b] into xf (f32) and xh (f16) ----
    {
        const float* xb = x + (size_t)b * SEQ * DIM;
        for (int idx = tid; idx < SEQ * DIM; idx += 256) {
            int r = idx / DIM, d = idx - r * DIM;
            float v = xb[idx];
            xf[r * 40 + d] = v;
            xh[r * 72 + d] = (_Float16)v;
        }
    }
    __syncthreads();

    const float scale = 0.22941573f; // 1/sqrt(19)

    for (int layer = 0; layer < NLAY; ++layer) {
        // ---- stage this layer's weights into LDS ----
        {
            const _Float16* gq = ws + WS_QKV + layer * 8192;
            for (int c = tid; c < 1024; c += 256) {
                int row = c >> 3, ch = (c & 7) * 8;
                *(int4*)(wq + row * 72 + ch) = *(const int4*)(gq + row * 64 + ch);
            }
            const _Float16* g0 = ws + WS_OUT + layer * 4096;
            const _Float16* g1 = ws + WS_L1  + layer * 4096;
            const _Float16* g2 = ws + WS_L2  + layer * 4096;
            for (int c = tid; c < 512; c += 256) {
                int row = c >> 3, ch = (c & 7) * 8;
                *(int4*)(wsm + 0 * 4608 + row * 72 + ch) = *(const int4*)(g0 + row * 64 + ch);
                *(int4*)(wsm + 1 * 4608 + row * 72 + ch) = *(const int4*)(g1 + row * 64 + ch);
                *(int4*)(wsm + 2 * 4608 + row * 72 + ch) = *(const int4*)(g2 + row * 64 + ch);
            }
            // hint next layer's weights toward L2 (global_prefetch_b8)
            if (layer + 1 < NLAY && tid < 64)
                __builtin_prefetch(ws + WS_QKV + (layer + 1) * 8192 + tid * 128, 0, 1);
        }
        __syncthreads();

        // ---- QKV: [64x64pad] x [64 x 128pad] -> scatter q/k/vT (branchless dest) ----
        for (int t = wave; t < 32; t += 8) {
            int mt = t & 3, nt = t >> 2;
            v8f acc = {};
            #pragma unroll
            for (int ks = 0; ks < 2; ++ks) {
                v16h a  = load_frag(xh + (mt * 16 + lcol) * 72 + ks * 32 + lkof);
                v16h bf = load_frag(wq + (nt * 16 + lcol) * 72 + ks * 32 + lkof);
                acc = wmma16(a, bf, acc);
            }
            int n = nt * 16 + lcol;
            bool valid = n < N3;
            int nn  = valid ? n : 0;
            int sec = nn / DIM;              // 0=q, 1=k, 2=v
            int wi  = nn - sec * DIM;
            int h   = wi / HDIM;
            int f   = wi - h * HDIM;
            // branchless destination base + stride (cndmask, no exec ladder)
            _Float16* base01 = (sec == 0) ? qbuf : kbuf;
            _Float16* dst = (sec == 2) ? (vT + h * 2304 + f * 72)
                                       : (base01 + h * 2560 + f);
            int stride = (sec == 2) ? 1 : 40;
            dst += (mt * 16 + lhi) * stride;
            float bias = valid ? ipb[layer * N3 + nn] : 0.f;
            if (valid) {
                #pragma unroll
                for (int g = 0; g < 8; ++g)
                    dst[g * stride] = (_Float16)(acc[g] + bias);
            }
        }
        __syncthreads();

        // ---- scores: Q @ K^T per head -> sbuf (f32, scaled) ----
        for (int t = wave; t < 32; t += 8) {
            int h = t >> 4, tt = t & 15, mt = tt & 3, nt = tt >> 2;
            v8f acc = {};
            v16h a  = load_frag(qbuf + h * 2560 + (mt * 16 + lcol) * 40 + lkof);
            v16h bf = load_frag(kbuf + h * 2560 + (nt * 16 + lcol) * 40 + lkof);
            acc = wmma16(a, bf, acc);
            float* dst = sbuf + h * 4224 + (mt * 16 + lhi) * 66 + nt * 16 + lcol;
            #pragma unroll
            for (int g = 0; g < 8; ++g) dst[g * 66] = acc[g] * scale;
        }
        __syncthreads();

        // ---- row softmax -> pbuf (f16) ----
        if (tid < 128) {
            int h = tid >> 6, l = tid & 63;
            float* r = sbuf + h * 4224 + l * 66;
            float mx = r[0];
            for (int j = 1; j < SEQ; ++j) mx = fmaxf(mx, r[j]);
            float s = 0.f;
            for (int j = 0; j < SEQ; ++j) { float e = __expf(r[j] - mx); r[j] = e; s += e; }
            float inv = 1.f / s;
            _Float16* pr = pbuf + h * 4608 + l * 72;
            for (int j = 0; j < SEQ; ++j) pr[j] = (_Float16)(r[j] * inv);
        }
        __syncthreads();

        // ---- P @ V -> aob (concat heads) ----
        for (int t = wave; t < 16; t += 8) {
            int h = t >> 3, tt = t & 7, mt = tt & 3, nt = tt >> 2;
            v8f acc = {};
            #pragma unroll
            for (int ks = 0; ks < 2; ++ks) {
                v16h a  = load_frag(pbuf + h * 4608 + (mt * 16 + lcol) * 72 + ks * 32 + lkof);
                v16h bf = load_frag(vT   + h * 2304 + (nt * 16 + lcol) * 72 + ks * 32 + lkof);
                acc = wmma16(a, bf, acc);
            }
            int f = nt * 16 + lcol;
            if (f < HDIM) {
                _Float16* dst = aob + (mt * 16 + lhi) * 72 + h * HDIM + f;
                #pragma unroll
                for (int g = 0; g < 8; ++g) dst[g * 72] = (_Float16)acc[g];
            }
        }
        __syncthreads();

        // ---- out proj -> sbuf (f32 [64][40]) ----
        for (int t = wave; t < 16; t += 8) {
            int mt = t & 3, nt = t >> 2;
            v8f acc = {};
            #pragma unroll
            for (int ks = 0; ks < 2; ++ks) {
                v16h a  = load_frag(aob + (mt * 16 + lcol) * 72 + ks * 32 + lkof);
                v16h bf = load_frag(wsm + 0 * 4608 + (nt * 16 + lcol) * 72 + ks * 32 + lkof);
                acc = wmma16(a, bf, acc);
            }
            int n = nt * 16 + lcol;
            if (n < DIM) {
                float bias = outb[layer * DIM + n];
                float* dst = sbuf + (mt * 16 + lhi) * 40 + n;
                #pragma unroll
                for (int g = 0; g < 8; ++g) dst[g * 40] = acc[g] + bias;
            }
        }
        __syncthreads();

        // ---- residual + LayerNorm 1 -> xf, xh ----
        if (tid < SEQ) {
            int r = tid;
            float tmp[DIM];
            float mu = 0.f;
            #pragma unroll
            for (int d = 0; d < DIM; ++d) {
                float s = xf[r * 40 + d] + sbuf[r * 40 + d];
                tmp[d] = s; mu += s;
            }
            mu *= (1.f / DIM);
            float var = 0.f;
            #pragma unroll
            for (int d = 0; d < DIM; ++d) { float c = tmp[d] - mu; var += c * c; }
            var *= (1.f / DIM);
            float inv = rsqrtf(var + 1e-5f);
            #pragma unroll
            for (int d = 0; d < DIM; ++d) {
                float v = (tmp[d] - mu) * inv * ln1w[layer * DIM + d] + ln1b[layer * DIM + d];
                xf[r * 40 + d] = v;
                xh[r * 72 + d] = (_Float16)v;
            }
        }
        __syncthreads();

        // ---- lin1 + GELU(exact) -> pbuf reused as hidden [64][72] ----
        _Float16* hb = pbuf;
        for (int t = wave; t < 16; t += 8) {
            int mt = t & 3, nt = t >> 2;
            v8f acc = {};
            #pragma unroll
            for (int ks = 0; ks < 2; ++ks) {
                v16h a  = load_frag(xh + (mt * 16 + lcol) * 72 + ks * 32 + lkof);
                v16h bf = load_frag(wsm + 1 * 4608 + (nt * 16 + lcol) * 72 + ks * 32 + lkof);
                acc = wmma16(a, bf, acc);
            }
            int n = nt * 16 + lcol;
            float bias = (n < DIM) ? l1b[layer * DIM + n] : 0.f;
            _Float16* dst = hb + (mt * 16 + lhi) * 72 + n;
            #pragma unroll
            for (int g = 0; g < 8; ++g) {
                float v = acc[g] + bias;
                v = 0.5f * v * (1.f + erff(v * 0.70710678f));
                dst[g * 72] = (_Float16)v;   // n>=38 writes exact zeros (pad)
            }
        }
        __syncthreads();

        // ---- lin2 -> sbuf (f32 [64][40]) ----
        for (int t = wave; t < 16; t += 8) {
            int mt = t & 3, nt = t >> 2;
            v8f acc = {};
            #pragma unroll
            for (int ks = 0; ks < 2; ++ks) {
                v16h a  = load_frag(hb + (mt * 16 + lcol) * 72 + ks * 32 + lkof);
                v16h bf = load_frag(wsm + 2 * 4608 + (nt * 16 + lcol) * 72 + ks * 32 + lkof);
                acc = wmma16(a, bf, acc);
            }
            int n = nt * 16 + lcol;
            if (n < DIM) {
                float bias = l2b[layer * DIM + n];
                float* dst = sbuf + (mt * 16 + lhi) * 40 + n;
                #pragma unroll
                for (int g = 0; g < 8; ++g) dst[g * 40] = acc[g] + bias;
            }
        }
        __syncthreads();

        // ---- residual + LayerNorm 2 -> xf, xh ----
        if (tid < SEQ) {
            int r = tid;
            float tmp[DIM];
            float mu = 0.f;
            #pragma unroll
            for (int d = 0; d < DIM; ++d) {
                float s = xf[r * 40 + d] + sbuf[r * 40 + d];
                tmp[d] = s; mu += s;
            }
            mu *= (1.f / DIM);
            float var = 0.f;
            #pragma unroll
            for (int d = 0; d < DIM; ++d) { float c = tmp[d] - mu; var += c * c; }
            var *= (1.f / DIM);
            float inv = rsqrtf(var + 1e-5f);
            #pragma unroll
            for (int d = 0; d < DIM; ++d) {
                float v = (tmp[d] - mu) * inv * ln2w[layer * DIM + d] + ln2b[layer * DIM + d];
                xf[r * 40 + d] = v;
                xh[r * 72 + d] = (_Float16)v;
            }
        }
        __syncthreads();
    }

    // ================= pooling + classifier =================
    float* sp = sbuf;             // [8][66] raw/softmaxed scores
    float* fb = sbuf + 1024;      // [8][40] feats
    for (int idx = tid; idx < NK * SEQ; idx += 256) {
        int k = idx >> 6, l = idx & 63;
        float s = attnb[k];
        #pragma unroll
        for (int d = 0; d < DIM; ++d) s += xf[l * 40 + d] * attnw[k * DIM + d];
        sp[k * 66 + l] = s;
    }
    __syncthreads();

    if (tid < NK) {
        int k = tid;
        float* r = sp + k * 66;
        float mx = r[0];
        for (int j = 1; j < SEQ; ++j) mx = fmaxf(mx, r[j]);
        float s = 0.f;
        for (int j = 0; j < SEQ; ++j) { float e = __expf(r[j] - mx); r[j] = e; s += e; }
        float inv = 1.f / s;
        float* so = out + 65536 + ((size_t)b * NK + k) * SEQ;
        for (int j = 0; j < SEQ; ++j) { float p = r[j] * inv; r[j] = p; so[j] = p; }
    }
    __syncthreads();

    for (int idx = tid; idx < NK * DIM; idx += 256) {
        int k = idx / DIM, d = idx - k * DIM;
        float s = 0.f;
        for (int l = 0; l < SEQ; ++l) s += sp[k * 66 + l] * xf[l * 40 + d];
        fb[k * 40 + d] = s;
    }
    __syncthreads();

    if (tid < NK) {
        float s = clfb[tid];
        #pragma unroll
        for (int d = 0; d < DIM; ++d) s += fb[tid * 40 + d] * clfw[tid * DIM + d];
        out[(size_t)b * NK + tid] = s;
    }
}

extern "C" void kernel_launch(void* const* d_in, const int* in_sizes, int n_in,
                              void* d_out, int out_size, void* d_ws, size_t ws_size,
                              hipStream_t stream) {
    const float* x     = (const float*)d_in[0];
    const float* ipw   = (const float*)d_in[1];
    const float* ipb   = (const float*)d_in[2];
    const float* outw  = (const float*)d_in[3];
    const float* outb  = (const float*)d_in[4];
    const float* l1w   = (const float*)d_in[5];
    const float* l1b   = (const float*)d_in[6];
    const float* l2w   = (const float*)d_in[7];
    const float* l2b   = (const float*)d_in[8];
    const float* ln1w  = (const float*)d_in[9];
    const float* ln1b  = (const float*)d_in[10];
    const float* ln2w  = (const float*)d_in[11];
    const float* ln2b  = (const float*)d_in[12];
    const float* attnw = (const float*)d_in[13];
    const float* attnb = (const float*)d_in[14];
    const float* clfw  = (const float*)d_in[15];
    const float* clfb  = (const float*)d_in[16];

    _Float16* ws = (_Float16*)d_ws;
    float* out = (float*)d_out;

    cvt_weights_kernel<<<(WS_TOTAL + 255) / 256, 256, 0, stream>>>(ipw, outw, l1w, l2w, ws);
    symptom_fused_kernel<<<8192, 256, 0, stream>>>(
        x, ipb, outb, l1b, l2b, ln1w, ln1b, ln2w, ln2b,
        attnw, attnb, clfw, clfb, ws, out);
}
